// FeedForward_44822278701282
// MI455X (gfx1250) — matrix operand order repaired and checked by reading, not compiled, not run on hardware
//
#include <hip/hip_runtime.h>
#include <stdint.h>

#define DIM_   4096
#define HID_   11008
#define R_LORA 8
#define SCAL   4.0f
#define M_TOK  8192      // B*S = 4*2048

// ---- tile config ----
#define BM 128
#define BN 64
#define BK 64
#define LDA 72   // BK + 8 pad (keeps 16B alignment: 72*2 = 144 = 9*16)
#define LDB 72   // BN + 8 pad

typedef __bf16 bf16_t;
typedef __bf16 v16bf __attribute__((ext_vector_type(16)));
typedef float  v8f   __attribute__((ext_vector_type(8)));
typedef int    v4i   __attribute__((vector_size(4 * sizeof(int))));
typedef __attribute__((address_space(1))) v4i* gptr_v4i;
typedef __attribute__((address_space(3))) v4i* lptr_v4i;

static __device__ __forceinline__ bf16_t f2bf(float f) {
    union { float f; uint32_t u; } v; v.f = f;
    uint32_t r = (v.u + 0x7FFFu + ((v.u >> 16) & 1u)) >> 16;   // RNE
    union { uint16_t s; bf16_t b; } o; o.s = (uint16_t)r;
    return o.b;
}
static __device__ __forceinline__ float bf2f(bf16_t b) {
    union { uint16_t s; bf16_t b; } i; i.b = b;
    union { float f; uint32_t u; } v; v.u = ((uint32_t)i.s) << 16;
    return v.f;
}
static __device__ __forceinline__ float to_f(float x)  { return x; }
static __device__ __forceinline__ float to_f(bf16_t x) { return bf2f(x); }

// ---- CDNA5 async global->LDS 16B copy (ASYNCcnt-tracked, no VGPR round trip)
static __device__ __forceinline__ void async_cp16(bf16_t* lds, const bf16_t* g) {
#if __has_builtin(__builtin_amdgcn_global_load_async_to_lds_b128)
    __builtin_amdgcn_global_load_async_to_lds_b128(
        (gptr_v4i)(uintptr_t)g, (lptr_v4i)lds, 0, 0);
#else
    const uint32_t l =
        (uint32_t)(size_t)(__attribute__((address_space(3))) bf16_t*)lds;
    asm volatile("global_load_async_to_lds_b128 %0, %1, off"
                 :: "v"(l), "v"((const void*)g) : "memory");
#endif
}
static __device__ __forceinline__ void wait_async0() {
#if __has_builtin(__builtin_amdgcn_s_wait_asynccnt)
    __builtin_amdgcn_s_wait_asynccnt(0);
#else
    asm volatile("s_wait_asynccnt 0x0" ::: "memory");
#endif
}

// ---------------- fp32 -> bf16 bulk conversion (pre-pass) ------------------
__global__ __launch_bounds__(256) void cvt_bf16_kernel(
    const float* __restrict__ src, bf16_t* __restrict__ dst, int n4)
{
    const int i = blockIdx.x * 256 + threadIdx.x;
    if (i < n4) {
        const float4 v = reinterpret_cast<const float4*>(src)[i];
        union { bf16_t b[4]; uint2 u; } o;
        o.b[0] = f2bf(v.x); o.b[1] = f2bf(v.y); o.b[2] = f2bf(v.z); o.b[3] = f2bf(v.w);
        reinterpret_cast<uint2*>(dst)[i] = o.u;
    }
}

// ---------------- t = X @ A  (per-token rank-8 projections) ----------------
template <typename T>
__global__ __launch_bounds__(256) void lora_t_kernel(
    const T* __restrict__ X, int K,
    const float* __restrict__ A0, float* __restrict__ T0,
    const float* __restrict__ A1, float* __restrict__ T1)
{
    const int lane = threadIdx.x & 31;
    const int wave = threadIdx.x >> 5;
    const int m = blockIdx.x * 8 + wave;
    const T* xr = X + (size_t)m * K;

    float s0[R_LORA], s1[R_LORA];
#pragma unroll
    for (int r = 0; r < R_LORA; ++r) { s0[r] = 0.f; s1[r] = 0.f; }

    for (int k = lane; k < K; k += 32) {
        const float xv = to_f(xr[k]);
        const float* a0 = A0 + (size_t)k * R_LORA;
#pragma unroll
        for (int r = 0; r < R_LORA; ++r) s0[r] = fmaf(xv, a0[r], s0[r]);
        if (A1) {
            const float* a1 = A1 + (size_t)k * R_LORA;
#pragma unroll
            for (int r = 0; r < R_LORA; ++r) s1[r] = fmaf(xv, a1[r], s1[r]);
        }
    }
#pragma unroll
    for (int r = 0; r < R_LORA; ++r) {
        float v0 = s0[r], v1 = s1[r];
#pragma unroll
        for (int off = 16; off >= 1; off >>= 1) {
            v0 += __shfl_xor(v0, off, 32);
            v1 += __shfl_xor(v1, off, 32);
        }
        if (lane == 0) {
            T0[m * R_LORA + r] = v0;
            if (A1) T1[m * R_LORA + r] = v1;
        }
    }
}

// ---------------- GEMM1: h = silu(x@w1 + lora_g) * (x@w3 + lora_u) ---------
// PRE=true : X/W1/W3 are pre-converted bf16; staging = async copies
// PRE=false: X/W1/W3 are fp32; staging = load+cvt+ds_store
template <bool PRE>
__global__ __launch_bounds__(256) void gemm1_kernel(
    const void* __restrict__ Xv,
    const void* __restrict__ W1v,
    const void* __restrict__ W3v,
    const float* __restrict__ TW,
    const float* __restrict__ Tup,
    const float* __restrict__ Tgate,
    const float* __restrict__ Bup,
    const float* __restrict__ Bgate,
    bf16_t* __restrict__ H)
{
    __shared__ bf16_t Als[2][BM][LDA];
    __shared__ bf16_t B1ls[2][BK][LDB];
    __shared__ bf16_t B3ls[2][BK][LDB];

    const int tid  = threadIdx.x;
    const int lane = tid & 31;
    const int wave = tid >> 5;
    const int wm = wave >> 1;          // 0..3
    const int wn = wave & 1;           // 0..1
    const int m0 = blockIdx.y * BM;
    const int n0 = blockIdx.x * BN;

    const v8f zc = {};
    v8f ag[2][2], au[2][2];
#pragma unroll
    for (int mi = 0; mi < 2; ++mi)
#pragma unroll
        for (int ni = 0; ni < 2; ++ni) { ag[mi][ni] = zc; au[mi][ni] = zc; }

    const int srow = tid >> 4;          // 0..15
    const int scol = (tid & 15) << 2;   // 0..60 step 4
    const int ar   = tid >> 3;          // 0..31
    const int ac8  = (tid & 7) << 3;    // 0..56 step 8

    auto stage = [&](int b, int kk) {
        if constexpr (PRE) {
            const bf16_t* Xb  = (const bf16_t*)Xv;
            const bf16_t* W1b = (const bf16_t*)W1v;
            const bf16_t* W3b = (const bf16_t*)W3v;
#pragma unroll
            for (int p = 0; p < 4; ++p) {
                const int r = p * 32 + ar;
                async_cp16(&Als[b][r][ac8], Xb + (size_t)(m0 + r) * DIM_ + kk + ac8);
            }
#pragma unroll
            for (int p = 0; p < 2; ++p) {
                const int r = p * 32 + ar;
                async_cp16(&B1ls[b][r][ac8], W1b + (size_t)(kk + r) * HID_ + n0 + ac8);
                async_cp16(&B3ls[b][r][ac8], W3b + (size_t)(kk + r) * HID_ + n0 + ac8);
            }
        } else {
            const float* X  = (const float*)Xv;
            const float* W1 = (const float*)W1v;
            const float* W3 = (const float*)W3v;
#pragma unroll
            for (int p = 0; p < 8; ++p) {
                const int r = p * 16 + srow;
                const float4 v = *reinterpret_cast<const float4*>(
                    X + (size_t)(m0 + r) * DIM_ + kk + scol);
                bf16_t* d = &Als[b][r][scol];
                d[0] = f2bf(v.x); d[1] = f2bf(v.y); d[2] = f2bf(v.z); d[3] = f2bf(v.w);
            }
#pragma unroll
            for (int p = 0; p < 4; ++p) {
                const int r = p * 16 + srow;
                const size_t off = (size_t)(kk + r) * HID_ + n0 + scol;
                const float4 v1 = *reinterpret_cast<const float4*>(W1 + off);
                const float4 v3 = *reinterpret_cast<const float4*>(W3 + off);
                bf16_t* d1 = &B1ls[b][r][scol];
                d1[0] = f2bf(v1.x); d1[1] = f2bf(v1.y); d1[2] = f2bf(v1.z); d1[3] = f2bf(v1.w);
                bf16_t* d3 = &B3ls[b][r][scol];
                d3[0] = f2bf(v3.x); d3[1] = f2bf(v3.y); d3[2] = f2bf(v3.z); d3[3] = f2bf(v3.w);
            }
        }
    };

    constexpr int NT = DIM_ / BK;
    stage(0, 0);
    for (int it = 0; it < NT; ++it) {
        const int cb = it & 1;
        if constexpr (PRE) wait_async0();
        __syncthreads();
        if (it + 1 < NT) stage(cb ^ 1, (it + 1) * BK);

#pragma unroll
        for (int ks = 0; ks < BK; ks += 32) {
            v16bf afr[2], b1fr[2], b3fr[2];
#pragma unroll
            for (int mi = 0; mi < 2; ++mi) {
                const bf16_t* pa =
                    &Als[cb][wm * 32 + mi * 16 + (lane & 15)][ks + ((lane >> 4) << 3)];
                uint4* ap = reinterpret_cast<uint4*>(&afr[mi]);
                ap[0] = *reinterpret_cast<const uint4*>(pa);
                ap[1] = *reinterpret_cast<const uint4*>(pa + 16);
            }
#pragma unroll
            for (int ni = 0; ni < 2; ++ni) {
                const int nn = wn * 32 + ni * 16 + (lane & 15);
#pragma unroll
                for (int s = 0; s < 16; ++s) {
                    const int kr = ks + ((lane >> 4) << 3) + (s & 7) + ((s >> 3) << 4);
                    b1fr[ni][s] = B1ls[cb][kr][nn];
                    b3fr[ni][s] = B3ls[cb][kr][nn];
                }
            }
#pragma unroll
            for (int mi = 0; mi < 2; ++mi)
#pragma unroll
                for (int ni = 0; ni < 2; ++ni) {
                    ag[mi][ni] = __builtin_amdgcn_wmma_f32_16x16x32_bf16(
                        false, afr[mi], false, b1fr[ni], (short)0, ag[mi][ni], false, false);
                    au[mi][ni] = __builtin_amdgcn_wmma_f32_16x16x32_bf16(
                        false, afr[mi], false, b3fr[ni], (short)0, au[mi][ni], false, false);
                }
        }
    }

    // epilogue: LoRA (rank 8) + SiLU * up, store bf16
#pragma unroll
    for (int ni = 0; ni < 2; ++ni) {
        const int n = n0 + wn * 32 + ni * 16 + (lane & 15);
        float bg[R_LORA], bu[R_LORA];
#pragma unroll
        for (int r = 0; r < R_LORA; ++r) {
            bg[r] = Bgate[r * HID_ + n];
            bu[r] = Bup[r * HID_ + n];
        }
#pragma unroll
        for (int mi = 0; mi < 2; ++mi) {
#pragma unroll
            for (int g = 0; g < 8; ++g) {
                const int m = m0 + wm * 32 + mi * 16 + ((lane >> 4) << 3) + g;
                const float* tg = Tgate + m * R_LORA;
                const float* tu = Tup + m * R_LORA;
                float lg = 0.f, lu = 0.f;
#pragma unroll
                for (int r = 0; r < R_LORA; ++r) {
                    lg = fmaf(tg[r], bg[r], lg);
                    lu = fmaf(tu[r], bu[r], lu);
                }
                const float gate = ag[mi][ni][g] + lg * SCAL * TW[m * 3 + 1];
                const float up   = au[mi][ni][g] + lu * SCAL * TW[m * 3 + 0];
                const float sg   = gate / (1.0f + __expf(-gate));
                H[(size_t)m * HID_ + n] = f2bf(sg * up);
            }
        }
    }
}

// ---------------- GEMM2: out = h@w2 + lora_down ----------------------------
// A (h) is always bf16 -> always async-staged. B follows PRE.
template <bool PRE>
__global__ __launch_bounds__(256) void gemm2_kernel(
    const bf16_t* __restrict__ H,
    const void* __restrict__ W2v,
    const float* __restrict__ TW,
    const float* __restrict__ Tdn,
    const float* __restrict__ Bdn,
    float* __restrict__ OUT)
{
    __shared__ bf16_t Als[2][BM][LDA];
    __shared__ bf16_t Bls[2][BK][LDB];

    const int tid  = threadIdx.x;
    const int lane = tid & 31;
    const int wave = tid >> 5;
    const int wm = wave >> 1;
    const int wn = wave & 1;
    const int m0 = blockIdx.y * BM;
    const int n0 = blockIdx.x * BN;

    const v8f zc = {};
    v8f acc[2][2];
#pragma unroll
    for (int mi = 0; mi < 2; ++mi)
#pragma unroll
        for (int ni = 0; ni < 2; ++ni) acc[mi][ni] = zc;

    const int srow = tid >> 4;
    const int scol = (tid & 15) << 2;
    const int ar   = tid >> 3;
    const int ac8  = (tid & 7) << 3;

    auto stage = [&](int b, int kk) {
        // A: h bf16, pure async copies
#pragma unroll
        for (int p = 0; p < 4; ++p) {
            const int r = p * 32 + ar;
            async_cp16(&Als[b][r][ac8], H + (size_t)(m0 + r) * HID_ + kk + ac8);
        }
        if constexpr (PRE) {
            const bf16_t* W2b = (const bf16_t*)W2v;
#pragma unroll
            for (int p = 0; p < 2; ++p) {
                const int r = p * 32 + ar;
                async_cp16(&Bls[b][r][ac8], W2b + (size_t)(kk + r) * DIM_ + n0 + ac8);
            }
        } else {
            const float* W2 = (const float*)W2v;
#pragma unroll
            for (int p = 0; p < 4; ++p) {
                const int r = p * 16 + srow;
                const float4 v = *reinterpret_cast<const float4*>(
                    W2 + (size_t)(kk + r) * DIM_ + n0 + scol);
                bf16_t* d = &Bls[b][r][scol];
                d[0] = f2bf(v.x); d[1] = f2bf(v.y); d[2] = f2bf(v.z); d[3] = f2bf(v.w);
            }
        }
    };

    constexpr int NT = HID_ / BK;
    stage(0, 0);
    for (int it = 0; it < NT; ++it) {
        const int cb = it & 1;
        wait_async0();                 // A staging is always async
        __syncthreads();
        if (it + 1 < NT) stage(cb ^ 1, (it + 1) * BK);

#pragma unroll
        for (int ks = 0; ks < BK; ks += 32) {
            v16bf afr[2], bfr[2];
#pragma unroll
            for (int mi = 0; mi < 2; ++mi) {
                const bf16_t* pa =
                    &Als[cb][wm * 32 + mi * 16 + (lane & 15)][ks + ((lane >> 4) << 3)];
                uint4* ap = reinterpret_cast<uint4*>(&afr[mi]);
                ap[0] = *reinterpret_cast<const uint4*>(pa);
                ap[1] = *reinterpret_cast<const uint4*>(pa + 16);
            }
#pragma unroll
            for (int ni = 0; ni < 2; ++ni) {
                const int nn = wn * 32 + ni * 16 + (lane & 15);
#pragma unroll
                for (int s = 0; s < 16; ++s) {
                    const int kr = ks + ((lane >> 4) << 3) + (s & 7) + ((s >> 3) << 4);
                    bfr[ni][s] = Bls[cb][kr][nn];
                }
            }
#pragma unroll
            for (int mi = 0; mi < 2; ++mi)
#pragma unroll
                for (int ni = 0; ni < 2; ++ni)
                    acc[mi][ni] = __builtin_amdgcn_wmma_f32_16x16x32_bf16(
                        false, afr[mi], false, bfr[ni], (short)0, acc[mi][ni], false, false);
        }
    }

#pragma unroll
    for (int ni = 0; ni < 2; ++ni) {
        const int n = n0 + wn * 32 + ni * 16 + (lane & 15);
        float bd[R_LORA];
#pragma unroll
        for (int r = 0; r < R_LORA; ++r) bd[r] = Bdn[r * DIM_ + n];
#pragma unroll
        for (int mi = 0; mi < 2; ++mi) {
#pragma unroll
            for (int g = 0; g < 8; ++g) {
                const int m = m0 + wm * 32 + mi * 16 + ((lane >> 4) << 3) + g;
                const float* td = Tdn + m * R_LORA;
                float ld = 0.f;
#pragma unroll
                for (int r = 0; r < R_LORA; ++r) ld = fmaf(td[r], bd[r], ld);
                OUT[(size_t)m * DIM_ + n] = acc[mi][ni][g] + ld * SCAL * TW[m * 3 + 2];
            }
        }
    }
}

extern "C" void kernel_launch(void* const* d_in, const int* in_sizes, int n_in,
                              void* d_out, int out_size, void* d_ws, size_t ws_size,
                              hipStream_t stream) {
    (void)in_sizes; (void)n_in; (void)out_size;
    const float* x      = (const float*)d_in[0];
    const float* tw     = (const float*)d_in[1];
    const float* w1     = (const float*)d_in[2];
    const float* w3     = (const float*)d_in[3];
    const float* w2     = (const float*)d_in[4];
    const float* a_up   = (const float*)d_in[5];
    const float* b_up   = (const float*)d_in[6];
    const float* a_gate = (const float*)d_in[7];
    const float* b_gate = (const float*)d_in[8];
    const float* a_down = (const float*)d_in[9];
    const float* b_down = (const float*)d_in[10];
    float* out = (float*)d_out;

    char* ws = (char*)d_ws;
    size_t off = 0;
    auto carve = [&](size_t bytes) -> char* {
        char* p = ws + off;
        off = (off + bytes + 255) & ~(size_t)255;
        return p;
    };
    bf16_t* H    = (bf16_t*)carve((size_t)M_TOK * HID_ * sizeof(bf16_t));
    float* Tup   = (float*)carve((size_t)M_TOK * R_LORA * sizeof(float));
    float* Tgate = (float*)carve((size_t)M_TOK * R_LORA * sizeof(float));
    float* Tdn   = (float*)carve((size_t)M_TOK * R_LORA * sizeof(float));
    // optional pre-converted bf16 copies
    bf16_t* Xb  = (bf16_t*)carve((size_t)M_TOK * DIM_ * sizeof(bf16_t));
    bf16_t* W1b = (bf16_t*)carve((size_t)DIM_ * HID_ * sizeof(bf16_t));
    bf16_t* W3b = (bf16_t*)carve((size_t)DIM_ * HID_ * sizeof(bf16_t));
    bf16_t* W2b = (bf16_t*)carve((size_t)HID_ * DIM_ * sizeof(bf16_t));
    const bool pre = ws_size >= off;   // deterministic per session

    if (pre) {
        const int nx = M_TOK * DIM_ / 4, nw = DIM_ * HID_ / 4;
        cvt_bf16_kernel<<<(nx + 255) / 256, 256, 0, stream>>>(x, Xb, nx);
        cvt_bf16_kernel<<<(nw + 255) / 256, 256, 0, stream>>>(w1, W1b, nw);
        cvt_bf16_kernel<<<(nw + 255) / 256, 256, 0, stream>>>(w3, W3b, nw);
        cvt_bf16_kernel<<<(nw + 255) / 256, 256, 0, stream>>>(w2, W2b, nw);
    }

    // 1) per-token rank-8 projections of x
    lora_t_kernel<float><<<M_TOK / 8, 256, 0, stream>>>(x, DIM_, a_up, Tup, a_gate, Tgate);
    // 2) fused gate/up GEMM + LoRA + SiLU -> h (bf16)
    if (pre)
        gemm1_kernel<true><<<dim3(HID_ / BN, M_TOK / BM), 256, 0, stream>>>(
            Xb, W1b, W3b, tw, Tup, Tgate, b_up, b_gate, H);
    else
        gemm1_kernel<false><<<dim3(HID_ / BN, M_TOK / BM), 256, 0, stream>>>(
            x, w1, w3, tw, Tup, Tgate, b_up, b_gate, H);
    // 3) per-token rank-8 projection of h
    lora_t_kernel<bf16_t><<<M_TOK / 8, 256, 0, stream>>>(H, HID_, a_down, Tdn,
                                                         (const float*)nullptr, (float*)nullptr);
    // 4) down GEMM + LoRA -> out (fp32)
    if (pre)
        gemm2_kernel<true><<<dim3(DIM_ / BN, M_TOK / BM), 256, 0, stream>>>(
            H, W2b, tw, Tdn, b_down, out);
    else
        gemm2_kernel<false><<<dim3(DIM_ / BN, M_TOK / BM), 256, 0, stream>>>(
            H, w2, tw, Tdn, b_down, out);
}
